// Attention_73048803770397
// MI455X (gfx1250) — matrix operand order repaired and checked by reading, not compiled, not run on hardware
//
#include <hip/hip_runtime.h>
#include <hip/hip_bf16.h>

typedef _Float16 f16;
typedef __attribute__((ext_vector_type(16))) _Float16 v16h;
typedef __attribute__((ext_vector_type(8)))  float    v8f;

#define TT 2048
#define CC 1024
#define HH 16
#define DD 64
#define ASTR 40    // padded LDS stride (halves) for A tile rows (avoid bank conflicts)
#define BSTR 136   // padded LDS stride (halves) for B tile rows

// ---------------- elementwise helpers ----------------

__global__ void cast_f32_to_f16(const float* __restrict__ in, f16* __restrict__ out, int n) {
    int i = blockIdx.x * blockDim.x + threadIdx.x;
    if (i < n) out[i] = (f16)in[i];
}

// wt[k][n] = w[n][k]  (1024x1024), cast to f16
__global__ void transpose_w(const float* __restrict__ w, f16* __restrict__ wt) {
    int i = blockIdx.x * blockDim.x + threadIdx.x;
    int k = i >> 10, n = i & 1023;
    wt[i] = (f16)w[n * CC + k];
}

// kt[((b*H+h)*D+d)*T + t] = kh[(b*T+t)*C + h*D + d]
__global__ void transpose_k(const f16* __restrict__ kh, f16* __restrict__ kt) {
    int i  = blockIdx.x * blockDim.x + threadIdx.x;
    int t  = i & (TT - 1);
    int r  = i >> 11;
    int d  = r & (DD - 1);
    int bh = r >> 6;
    int h  = bh & (HH - 1);
    int b  = bh >> 4;
    kt[i] = kh[(size_t)(b * TT + t) * CC + h * DD + d];
}

// issue one async global->LDS 128-bit copy (per-lane addresses), ASYNCcnt-tracked
__device__ __forceinline__ void async_copy_b128(const f16* gsrc, f16* ldst) {
    unsigned long long ga = (unsigned long long)(uintptr_t)gsrc;
    unsigned int       la = (unsigned int)(uintptr_t)ldst;  // generic->LDS offset (low 32 bits)
    asm volatile("global_load_async_to_lds_b128 %0, %1, off"
                 :: "v"(la), "v"(ga) : "memory");
}

// ---------------- WMMA GEMM: Y[M,1024] = A[M,1024] * BT[1024,1024] ----------------
// BT is pre-transposed (K-major rows), so both operands stream contiguously.
// Workgroup: 256 threads = 8 waves; tile 128(M) x 128(N); wave = 32x64 (2x4 frags).
// LDS double-buffered, filled with GLOBAL_LOAD_ASYNC_TO_LDS_B128: stage k+1 streams
// in behind the 8-WMMA compute of stage k; s_wait_asynccnt retires stages in order.

template <bool OUT_F32>
__global__ void __launch_bounds__(256)
gemm128(const f16* __restrict__ A, const f16* __restrict__ BT, void* __restrict__ out) {
    __shared__ f16 As[2][128 * ASTR];
    __shared__ f16 Bs[2][32 * BSTR];

    const int tid  = threadIdx.x;
    const int lane = tid & 31;
    const int wave = tid >> 5;
    const int m0   = blockIdx.y * 128;
    const int n0   = blockIdx.x * 128;
    const int wm   = (wave & 3) * 32;
    const int wn   = (wave >> 2) * 64;
    const int mlo  = lane & 15;
    const int hi   = lane >> 4;

    // staging coordinates: 4 x b128 per thread per stage (A 128x32, B 32x128)
    const int c0 = tid, c1 = tid + 256;
    const int rA0 = c0 >> 2, gA0 = c0 & 3;
    const int rA1 = c1 >> 2, gA1 = c1 & 3;
    const int rB0 = c0 >> 4, gB0 = c0 & 15;
    const int rB1 = c1 >> 4, gB1 = c1 & 15;

    v8f acc[2][4];
#pragma unroll
    for (int mt = 0; mt < 2; ++mt)
#pragma unroll
        for (int nt = 0; nt < 4; ++nt) acc[mt][nt] = (v8f)0.0f;

#define ISSUE_STAGE(buf, k0)                                                              \
    do {                                                                                  \
        async_copy_b128(&A[(size_t)(m0 + rA0) * 1024 + (k0) + gA0 * 8],                   \
                        &As[buf][rA0 * ASTR + gA0 * 8]);                                  \
        async_copy_b128(&A[(size_t)(m0 + rA1) * 1024 + (k0) + gA1 * 8],                   \
                        &As[buf][rA1 * ASTR + gA1 * 8]);                                  \
        async_copy_b128(&BT[(size_t)((k0) + rB0) * 1024 + n0 + gB0 * 8],                  \
                        &Bs[buf][rB0 * BSTR + gB0 * 8]);                                  \
        async_copy_b128(&BT[(size_t)((k0) + rB1) * 1024 + n0 + gB1 * 8],                  \
                        &Bs[buf][rB1 * BSTR + gB1 * 8]);                                  \
    } while (0)

    ISSUE_STAGE(0, 0);

    for (int k0 = 0; k0 < 1024; k0 += 32) {
        const int buf = (k0 >> 5) & 1;
        if (k0 + 32 < 1024) {
            ISSUE_STAGE(buf ^ 1, k0 + 32);
            // 4 newest (stage k+1) may remain outstanding; stage k is complete (in-order)
            asm volatile("s_wait_asynccnt 0x4" ::: "memory");
        } else {
            asm volatile("s_wait_asynccnt 0x0" ::: "memory");
        }
        __syncthreads();  // whole tile (all waves' copies) visible

        v16h a[2], b[4];
#pragma unroll
        for (int mt = 0; mt < 2; ++mt) {
            const f16* p = &As[buf][(wm + mt * 16 + mlo) * ASTR];
            *((uint4*)&a[mt] + 0) = *(const uint4*)(p + hi * 8);
            *((uint4*)&a[mt] + 1) = *(const uint4*)(p + 16 + hi * 8);
        }
#pragma unroll
        for (int nt = 0; nt < 4; ++nt) {
            const f16* p = &Bs[buf][wn + nt * 16 + mlo];
#pragma unroll
            for (int s = 0; s < 16; ++s)
                b[nt][s] = p[((s < 8) ? (hi * 8 + s) : (16 + hi * 8 + s - 8)) * BSTR];
        }
#pragma unroll
        for (int mt = 0; mt < 2; ++mt)
#pragma unroll
            for (int nt = 0; nt < 4; ++nt)
                acc[mt][nt] = __builtin_amdgcn_wmma_f32_16x16x32_f16(
                    false, a[mt], false, b[nt], (short)0, acc[mt][nt], false, false);
        __syncthreads();  // all waves done reading buf before it is re-staged
    }
#undef ISSUE_STAGE

#pragma unroll
    for (int mt = 0; mt < 2; ++mt)
#pragma unroll
        for (int nt = 0; nt < 4; ++nt)
#pragma unroll
            for (int j = 0; j < 8; ++j) {
                size_t m = m0 + wm + mt * 16 + hi * 8 + j;
                size_t n = n0 + wn + nt * 16 + mlo;
                float  v = acc[mt][nt][j];
                if (OUT_F32) ((float*)out)[m * 1024 + n] = v;
                else         ((f16*)out)[m * 1024 + n]   = (f16)v;
            }
}

// ---------------- flash attention: 1 wave per (b, h, 16-query tile) ----------------

__global__ void __launch_bounds__(32)
attn_kernel(const f16* __restrict__ qh, const f16* __restrict__ kt,
            const f16* __restrict__ vh, f16* __restrict__ ctx) {
    __shared__ f16 Ps[16 * ASTR];

    const int lane = threadIdx.x;
    const int mlo  = lane & 15;
    const int hi   = lane >> 4;
    const int q0   = blockIdx.x * 16;
    const int h    = blockIdx.y;
    const int b    = blockIdx.z;

    // Q fragments over head-dim (gemm-K = 64 -> two 16x32 A-frags)
    v16h qa[2];
#pragma unroll
    for (int s = 0; s < 2; ++s) {
        const f16* p = &qh[(size_t)(b * TT + q0 + mlo) * CC + h * DD + s * 32];
        *((uint4*)&qa[s] + 0) = *(const uint4*)(p + hi * 8);
        *((uint4*)&qa[s] + 1) = *(const uint4*)(p + 16 + hi * 8);
    }

    v8f o[4];
#pragma unroll
    for (int t = 0; t < 4; ++t) o[t] = (v8f)0.0f;
    float rmax[8], rsum[8];
#pragma unroll
    for (int j = 0; j < 8; ++j) { rmax[j] = -3.0e30f; rsum[j] = 0.0f; }

    const int kbmax = (q0 + 47) >> 5;  // causal: key blocks covering keys <= q0+15
    for (int kb = 0; kb < kbmax; ++kb) {
        // S = Q * K^T  (two 16x16 key frags, accumulate over head-dim)
        v8f s01[2];
        s01[0] = (v8f)0.0f; s01[1] = (v8f)0.0f;
#pragma unroll
        for (int ds = 0; ds < 2; ++ds) {
#pragma unroll
            for (int f = 0; f < 2; ++f) {
                v16h bk;
                const f16* p = &kt[((size_t)(b * HH + h) * DD + ds * 32) * TT +
                                   kb * 32 + f * 16 + mlo];
#pragma unroll
                for (int s = 0; s < 16; ++s)
                    bk[s] = p[(size_t)((s < 8) ? (hi * 8 + s) : (16 + hi * 8 + s - 8)) * TT];
                s01[f] = __builtin_amdgcn_wmma_f32_16x16x32_f16(
                    false, qa[ds], false, bk, (short)0, s01[f], false, false);
            }
        }

        // scale + causal mask + online softmax (stats replicated per 16-lane half)
        float corr[8];
#pragma unroll
        for (int j = 0; j < 8; ++j) {
            int   qabs = q0 + hi * 8 + j;
            float v0 = s01[0][j] * 0.125f;
            float v1 = s01[1][j] * 0.125f;
            v0 = (kb * 32 + mlo      > qabs) ? -3.0e30f : v0;
            v1 = (kb * 32 + 16 + mlo > qabs) ? -3.0e30f : v1;
            float mx = fmaxf(v0, v1);
            mx = fmaxf(mx, __shfl_xor(mx, 1, 32));
            mx = fmaxf(mx, __shfl_xor(mx, 2, 32));
            mx = fmaxf(mx, __shfl_xor(mx, 4, 32));
            mx = fmaxf(mx, __shfl_xor(mx, 8, 32));
            float nm = fmaxf(rmax[j], mx);
            float c  = __expf(rmax[j] - nm);
            float p0 = __expf(v0 - nm);
            float p1 = __expf(v1 - nm);
            float ps = p0 + p1;
            ps += __shfl_xor(ps, 1, 32);
            ps += __shfl_xor(ps, 2, 32);
            ps += __shfl_xor(ps, 4, 32);
            ps += __shfl_xor(ps, 8, 32);
            rsum[j] = rsum[j] * c + ps;
            rmax[j] = nm;
            corr[j] = c;
            Ps[(hi * 8 + j) * ASTR + mlo]      = (f16)p0;
            Ps[(hi * 8 + j) * ASTR + 16 + mlo] = (f16)p1;
        }
#pragma unroll
        for (int t = 0; t < 4; ++t)
#pragma unroll
            for (int j = 0; j < 8; ++j) o[t][j] *= corr[j];
        __syncthreads();  // single-wave WG: S_NOP, but orders LDS for the compiler

        // P (C-layout) -> A-fragment via LDS, then O += P * V
        v16h pa;
        {
            const f16* pp = &Ps[mlo * ASTR];
            *((uint4*)&pa + 0) = *(const uint4*)(pp + hi * 8);
            *((uint4*)&pa + 1) = *(const uint4*)(pp + 16 + hi * 8);
        }
#pragma unroll
        for (int t = 0; t < 4; ++t) {
            v16h vb;
            const f16* p = &vh[(size_t)(b * TT + kb * 32) * CC + h * DD + t * 16 + mlo];
#pragma unroll
            for (int s = 0; s < 16; ++s)
                vb[s] = p[(size_t)((s < 8) ? (hi * 8 + s) : (16 + hi * 8 + s - 8)) * CC];
            o[t] = __builtin_amdgcn_wmma_f32_16x16x32_f16(
                false, pa, false, vb, (short)0, o[t], false, false);
        }
        __syncthreads();
    }

    // normalize and emit context in f16 [B,T,C] (head-interleaved)
#pragma unroll
    for (int t = 0; t < 4; ++t)
#pragma unroll
        for (int j = 0; j < 8; ++j) {
            float v = o[t][j] / rsum[j];
            ctx[(size_t)(b * TT + q0 + hi * 8 + j) * CC + h * DD + t * 16 + mlo] = (f16)v;
        }
}

// ---------------- launch ----------------

extern "C" void kernel_launch(void* const* d_in, const int* in_sizes, int n_in,
                              void* d_out, int out_size, void* d_ws, size_t ws_size,
                              hipStream_t stream) {
    const float* x  = (const float*)d_in[0];
    // d_in[1] is the causal mask; we apply causality analytically.
    const float* wq = (const float*)d_in[2];
    const float* wk = (const float*)d_in[3];
    const float* wv = (const float*)d_in[4];
    const float* wo = (const float*)d_in[5];

    const size_t NX = (size_t)4 * TT * CC;  // 8,388,608
    const size_t NW = (size_t)CC * CC;      // 1,048,576

    f16* ws  = (f16*)d_ws;
    f16* xh  = ws; ws += NX;
    f16* wqT = ws; ws += NW;
    f16* wkT = ws; ws += NW;
    f16* wvT = ws; ws += NW;
    f16* woT = ws; ws += NW;
    f16* qh  = ws; ws += NX;
    f16* kh  = ws; ws += NX;
    f16* vh  = ws; ws += NX;
    f16* ktr = ws; ws += NX;
    f16* ctx = ws; ws += NX;

    cast_f32_to_f16<<<dim3((unsigned)(NX / 256)), dim3(256), 0, stream>>>(x, xh, (int)NX);
    transpose_w<<<dim3((unsigned)(NW / 256)), dim3(256), 0, stream>>>(wq, wqT);
    transpose_w<<<dim3((unsigned)(NW / 256)), dim3(256), 0, stream>>>(wk, wkT);
    transpose_w<<<dim3((unsigned)(NW / 256)), dim3(256), 0, stream>>>(wv, wvT);
    transpose_w<<<dim3((unsigned)(NW / 256)), dim3(256), 0, stream>>>(wo, woT);

    dim3 gg(8, 64), gb(256);
    gemm128<false><<<gg, gb, 0, stream>>>(xh, wqT, qh);
    gemm128<false><<<gg, gb, 0, stream>>>(xh, wkT, kh);
    gemm128<false><<<gg, gb, 0, stream>>>(xh, wvT, vh);

    transpose_k<<<dim3((unsigned)(NX / 256)), dim3(256), 0, stream>>>(kh, ktr);

    attn_kernel<<<dim3(TT / 16, HH, 4), dim3(32), 0, stream>>>(qh, ktr, vh, ctx);

    gemm128<true><<<gg, gb, 0, stream>>>(ctx, woT, d_out);
}
